// LoRALinear_1992864825716
// MI455X (gfx1250) — matrix operand order repaired and checked by reading, not compiled, not run on hardware
//
#include <hip/hip_runtime.h>

// ---------------------------------------------------------------------------
// LoRA linear forward for gfx1250 (MI455X):
//   out = x @ W^T + SCALE * ((x*mask) @ A) @ B
// Kernel 1: rank reduction t = SCALE*(x*mask)@A  -> bf16 [M,32] in workspace
// Kernel 2: tiled bf16 WMMA GEMM x@W^T, double-buffered LDS (1 barrier/stage,
//           TK=64 -> 16 WMMAs between barriers), LoRA epilogue = 1 extra WMMA
//           per 16x16 C tile (R=32 is exactly one bf16 WMMA K-step).
// ---------------------------------------------------------------------------

typedef __attribute__((ext_vector_type(16))) __bf16 v16bf;
typedef __attribute__((ext_vector_type(8)))  __bf16 bf16x8;  // 16 bytes
typedef __attribute__((ext_vector_type(4)))  __bf16 bf16x4;  // 8 bytes
typedef __attribute__((ext_vector_type(8)))  float  v8f;
typedef __attribute__((ext_vector_type(4)))  float  f32x4;

#define TM   128   // block tile M
#define TN   128   // block tile N
#define TK   64    // K stage (two bf16 WMMA K-steps)
#define ROWB 144   // LDS row stride bytes: 64 bf16 (128B) + 16B pad (conflict-free)

union FragU {
  v16bf  v;
  bf16x8 h[2];
};

__device__ __forceinline__ bf16x4 cvt_bf4(f32x4 v) {
  bf16x4 r;
  r.x = (__bf16)v.x; r.y = (__bf16)v.y; r.z = (__bf16)v.z; r.w = (__bf16)v.w;
  return r;
}

// ---------------------------------------------------------------------------
// Kernel 1: t[m, r] = scale * sum_k x[m,k] * mask[m,k] * A[k,r]
// One wave per row m; lane == rank r (R == 32 == wave32 width).
// ---------------------------------------------------------------------------
__global__ __launch_bounds__(256)
void lora_rank_kernel(const float* __restrict__ x, const float* __restrict__ mask,
                      const float* __restrict__ A, __bf16* __restrict__ t,
                      int M, int K, float scale) {
  const int lane = threadIdx.x & 31;
  const int wave = threadIdx.x >> 5;
  const int m = blockIdx.x * 8 + wave;
  if (m >= M) return;

  const float* xr = x    + (size_t)m * K;
  const float* mr = mask + (size_t)m * K;

  float acc = 0.0f;
  for (int k0 = 0; k0 < K; k0 += 32) {
    float xv = xr[k0 + lane] * mr[k0 + lane];
#pragma unroll
    for (int j = 0; j < 32; ++j) {
      float xj = __shfl(xv, j, 32);
      acc = fmaf(xj, A[(size_t)(k0 + j) * 32 + lane], acc);
    }
  }
  t[(size_t)m * 32 + lane] = (__bf16)(acc * scale);
}

// ---------------------------------------------------------------------------
// Kernel 2: out[m,n] = sum_k x[m,k]*W[n,k]  + LoRA epilogue.
// Block 256 threads = 8 waves (wave32); block tile 128x128; wave tile 32x64.
// Double-buffered LDS ping-pong: stage s computes from buf[s&1] while the
// prefetch registers for stage s+1 are drained into buf[(s+1)&1]; one
// workgroup barrier per 64-wide K stage.
// ---------------------------------------------------------------------------
__global__ __launch_bounds__(256)
void lora_gemm_kernel(const float* __restrict__ x, const float* __restrict__ W,
                      const float* __restrict__ Bm, const __bf16* __restrict__ t,
                      float* __restrict__ out, int N, int K) {
  __shared__ __align__(16) char xsm[2][TM * ROWB];   // 2 x 18432 B
  __shared__ __align__(16) char wsm[2][TN * ROWB];   // 2 x 18432 B

  const int tid  = threadIdx.x;
  const int lane = tid & 31;
  const int wave = tid >> 5;
  const int wm   = (wave & 3) * 32;  // wave row offset within block tile
  const int wn   = (wave >> 2) * 64; // wave col offset within block tile
  const int bm   = blockIdx.y * TM;
  const int bn   = blockIdx.x * TN;

  const int half = lane >> 4;        // lane half selects K striping (ISA layout)
  const int l16  = lane & 15;

  v8f acc[2][4] = {};

  // Register prefetch: 128x64 fp32 per matrix per stage = 8 float4/thread each
  f32x4 xg[8], wg[8];
  auto gload = [&](int k0) {
#pragma unroll
    for (int i = 0; i < 8; ++i) {
      const int f = tid + i * 256;   // float4 id within 128x64 tile
      const int row = f >> 4;        // 16 float4 per 64-element row
      const int seg = f & 15;
      xg[i] = *(const f32x4*)(x + (size_t)(bm + row) * K + k0 + seg * 4);
      wg[i] = *(const f32x4*)(W + (size_t)(bn + row) * K + k0 + seg * 4);
    }
  };
  auto lstore = [&](int buf) {
#pragma unroll
    for (int i = 0; i < 8; ++i) {
      const int f = tid + i * 256;
      const int row = f >> 4;
      const int seg = f & 15;
      *(bf16x4*)(xsm[buf] + row * ROWB + seg * 8) = cvt_bf4(xg[i]);
      *(bf16x4*)(wsm[buf] + row * ROWB + seg * 8) = cvt_bf4(wg[i]);
    }
  };

  const int nst = K / TK;            // 64 stages
  gload(0);
  lstore(0);
  if (nst > 1) gload(TK);
  __syncthreads();

  for (int s = 0; s < nst; ++s) {
    // Drain prefetch regs into the other buffer, then issue next global loads
    // so their latency overlaps the 16 WMMAs below.
    if (s + 1 < nst) lstore((s + 1) & 1);
    if (s + 2 < nst) gload((s + 2) * TK);

    const char* xb = xsm[s & 1];
    const char* wb = wsm[s & 1];
#pragma unroll
    for (int kk = 0; kk < 2; ++kk) {   // two 32-wide WMMA K-steps per stage
      // A fragments (16x32 bf16): VGPR0-3 -> K [8h,8h+8), VGPR4-7 -> K [16+8h,24+8h)
      FragU af[2];
#pragma unroll
      for (int mi = 0; mi < 2; ++mi) {
        const char* p = xb + (wm + mi * 16 + l16) * ROWB + kk * 64 + half * 16;
        af[mi].h[0] = *(const bf16x8*)(p);
        af[mi].h[1] = *(const bf16x8*)(p + 32);
      }
      // B fragments (32x16 bf16): VGPR0-3 -> K [8h,8h+8), VGPR4-7 -> K [16+8h,24+8h)
      FragU bfr[4];
#pragma unroll
      for (int ni = 0; ni < 4; ++ni) {
        const char* p = wb + (wn + ni * 16 + l16) * ROWB + kk * 64 + half * 16;
        bfr[ni].h[0] = *(const bf16x8*)(p);
        bfr[ni].h[1] = *(const bf16x8*)(p + 32);
      }
#pragma unroll
      for (int mi = 0; mi < 2; ++mi)
#pragma unroll
        for (int ni = 0; ni < 4; ++ni)
          acc[mi][ni] = __builtin_amdgcn_wmma_f32_16x16x32_bf16(
              false, af[mi].v, false, bfr[ni].v, (short)0, acc[mi][ni], false, false);
    }
    __syncthreads();
  }

  // ---- LoRA epilogue: acc += t_tile(16x32) @ Bm_tile(32x16), one WMMA each ----
  FragU xaf[2];
#pragma unroll
  for (int mi = 0; mi < 2; ++mi) {
    const __bf16* p = t + (size_t)(bm + wm + mi * 16 + l16) * 32 + half * 8;
    xaf[mi].h[0] = *(const bf16x8*)(p);        // K = 8h .. 8h+7
    xaf[mi].h[1] = *(const bf16x8*)(p + 16);   // K = 16+8h .. 23+8h
  }
  FragU bmf[4];
#pragma unroll
  for (int ni = 0; ni < 4; ++ni) {
    const int n = bn + wn + ni * 16 + l16;
#pragma unroll
    for (int e = 0; e < 16; ++e) {
      const int k = (e >> 3) * 16 + half * 8 + (e & 7);  // B layout: same K striping as A
      bmf[ni].v[e] = (__bf16)Bm[(size_t)k * N + n];
    }
  }
#pragma unroll
  for (int mi = 0; mi < 2; ++mi)
#pragma unroll
    for (int ni = 0; ni < 4; ++ni)
      acc[mi][ni] = __builtin_amdgcn_wmma_f32_16x16x32_bf16(
          false, xaf[mi].v, false, bmf[ni].v, (short)0, acc[mi][ni], false, false);

  // ---- Store C: VGPR v -> M = v + 8*half, N = lane&15 ----
#pragma unroll
  for (int mi = 0; mi < 2; ++mi) {
#pragma unroll
    for (int ni = 0; ni < 4; ++ni) {
      const int n = bn + wn + ni * 16 + l16;
#pragma unroll
      for (int v = 0; v < 8; ++v) {
        const int m = bm + wm + mi * 16 + v + half * 8;
        out[(size_t)m * N + n] = acc[mi][ni][v];
      }
    }
  }
}

// ---------------------------------------------------------------------------
// Launch wrapper
// Inputs (setup_inputs order): x[4,2048,4096], W[4096,4096], A[4096,32],
//                              B[32,4096], drop_mask[4,2048,4096]  (all fp32)
// ---------------------------------------------------------------------------
extern "C" void kernel_launch(void* const* d_in, const int* in_sizes, int n_in,
                              void* d_out, int out_size, void* d_ws, size_t ws_size,
                              hipStream_t stream) {
  const float* x    = (const float*)d_in[0];
  const float* W    = (const float*)d_in[1];
  const float* A    = (const float*)d_in[2];
  const float* Bm   = (const float*)d_in[3];
  const float* mask = (const float*)d_in[4];
  float* out = (float*)d_out;

  const int R = 32;
  const int K = in_sizes[2] / R;   // 4096
  const int N = in_sizes[3] / R;   // 4096
  const int M = in_sizes[0] / K;   // 8192

  __bf16* t = (__bf16*)d_ws;       // M*32 bf16 = 512 KB scratch
  const float SCALE = 32.0f / 32.0f;

  lora_rank_kernel<<<(M + 7) / 8, 256, 0, stream>>>(x, mask, A, t, M, K, SCALE);

  dim3 grid(N / TN, M / TM);       // 32 x 64 blocks
  lora_gemm_kernel<<<grid, 256, 0, stream>>>(x, W, Bm, t, out, N, K);
}
